// cross_module_29996051595336
// MI455X (gfx1250) — matrix operand order repaired and checked by reading, not compiled, not run on hardware
//
#include <hip/hip_runtime.h>

typedef _Float16 half16_t __attribute__((ext_vector_type(16)));
typedef _Float16 half8_t  __attribute__((ext_vector_type(8)));
typedef __fp16   fp16x2_t __attribute__((ext_vector_type(2)));   // cvt_pkrtz return type
typedef float    float8_t __attribute__((ext_vector_type(8)));
typedef float    float4_t __attribute__((ext_vector_type(4)));

#define BB   2
#define CC   64
#define C2C  32
#define NN   4096

// ---------------------------------------------------------------------------
// Kernel 1: 1x1-conv projections + channel reduction (max or mean) + v in f16
//   x: (B,64,N)   w*: (32,64)  b*: (32,)
//   qdst/kdst: (B,N) f32      vdst: (B,32,N) f16 (c-major so B-frags are rows)
// ---------------------------------------------------------------------------
__global__ __launch_bounds__(256) void proj_kernel(
    const float* __restrict__ x,
    const float* __restrict__ wq, const float* __restrict__ bq,
    const float* __restrict__ wk, const float* __restrict__ bk,
    const float* __restrict__ wv, const float* __restrict__ bv,
    float* __restrict__ qdst, float* __restrict__ kdst,
    _Float16* __restrict__ vdst, int do_mean)
{
    __shared__ float wq_s[C2C * CC], wk_s[C2C * CC], wv_s[C2C * CC];
    __shared__ float bq_s[C2C], bk_s[C2C], bv_s[C2C];

    const int tid = threadIdx.x;
    for (int i = tid; i < C2C * CC; i += 256) {
        wq_s[i] = wq[i]; wk_s[i] = wk[i]; wv_s[i] = wv[i];
    }
    if (tid < C2C) { bq_s[tid] = bq[tid]; bk_s[tid] = bk[tid]; bv_s[tid] = bv[tid]; }
    __syncthreads();

    const int n = blockIdx.x * 256 + tid;
    const int b = blockIdx.y;

    // coalesced: consecutive lanes read consecutive n
    const float* xp = x + (size_t)b * CC * NN + n;
    float xc[CC];
#pragma unroll
    for (int i = 0; i < CC; ++i) xc[i] = xp[(size_t)i * NN];

    float qred = do_mean ? 0.0f : -__builtin_inff();
    float kred = qred;
    _Float16* vp = vdst + (size_t)b * C2C * NN + n;

    for (int c2 = 0; c2 < C2C; ++c2) {
        const float* wqr = &wq_s[c2 * CC];
        const float* wkr = &wk_s[c2 * CC];
        const float* wvr = &wv_s[c2 * CC];
        float qa = bq_s[c2], ka = bk_s[c2], va = bv_s[c2];
#pragma unroll
        for (int i = 0; i < CC; ++i) {
            const float xv = xc[i];
            qa = __builtin_fmaf(wqr[i], xv, qa);
            ka = __builtin_fmaf(wkr[i], xv, ka);
            va = __builtin_fmaf(wvr[i], xv, va);
        }
        if (do_mean) { qred += qa; kred += ka; }
        else         { qred = fmaxf(qred, qa); kred = fmaxf(kred, ka); }
        vp[(size_t)c2 * NN] = (_Float16)va;
    }
    if (do_mean) { qred *= (1.0f / 32.0f); kred *= (1.0f / 32.0f); }
    qdst[(size_t)b * NN + n] = qred;
    kdst[(size_t)b * NN + n] = kred;
}

// ---------------------------------------------------------------------------
// Kernel 2: fused gate-attention GEMM (WMMA f16->f32) + 1x1 conv + residual
//   D[m,c] = sum_k gate(q[m]*kv[k]) * v[c,k];  out[o,m] = x[o,m] + g*(wc@D+b)
//   gate = 1/(1+exp2(sf * q * kv)), sf = +log2e (1-sigmoid) or -log2e (sigmoid)
// ---------------------------------------------------------------------------
__global__ __launch_bounds__(256) void attn_kernel(
    const float* __restrict__ qws, const float* __restrict__ kws,
    const _Float16* __restrict__ vws,
    const float* __restrict__ wc2, const float* __restrict__ bc2,
    const float* __restrict__ g1p, const float* __restrict__ x1,
    const float* __restrict__ wc1, const float* __restrict__ bc1,
    const float* __restrict__ g2p, const float* __restrict__ x2,
    float* __restrict__ out)
{
    __shared__ float kv_s[NN];                         // 16 KB
    __shared__ float wc_s[CC * C2C];                   //  8 KB
    __shared__ float bias_s[CC];
    __shared__ __align__(16) float mid_s[8][16][36];   // 18 KB (per-wave D tiles)

    const int br = blockIdx.y;     // 0: out1 (q1,k2,v1,wc2)  1: out2 (q2,k1,v2,wc1)
    const int b  = blockIdx.z;

    const float*    q    = qws + (size_t)(br * BB + b) * NN;
    const float*    kv   = kws + (size_t)(br * BB + b) * NN;
    const _Float16* v    = vws + (size_t)(br * BB + b) * C2C * NN;
    const float*    wc   = br ? wc1 : wc2;
    const float*    bias = br ? bc1 : bc2;
    const float*    xres = (br ? x2 : x1) + (size_t)b * CC * NN;
    const float     gamma = br ? g2p[0] : g1p[0];
    const float     sf    = br ? -1.44269504089f : 1.44269504089f;
    float* outp = out + (size_t)br * BB * CC * NN + (size_t)b * CC * NN;

    const int tid = threadIdx.x;
    for (int i = tid; i < NN; i += 256)       kv_s[i]  = kv[i];
    for (int i = tid; i < CC * C2C; i += 256) wc_s[i]  = wc[i];
    if (tid < CC)                             bias_s[tid] = bias[tid];
    __syncthreads();

    const int wave = tid >> 5;
    const int lane = tid & 31;
    const int ml   = lane & 15;      // m within tile (A/D rows) / c column
    const int kh   = lane >> 4;      // lane-half selector
    const int m0   = blockIdx.x * 128 + wave * 16;

    const float qf = q[m0 + ml] * sf;               // per-lane row scalar
    const _Float16* vp0 = v + (size_t)ml * NN;      // B frag, c = ml
    const _Float16* vp1 = v + (size_t)(ml + 16) * NN;

    float8_t acc0 = {}, acc1 = {};
    const float4_t* kv4 = (const float4_t*)kv_s;

#pragma unroll 2
    for (int kb = 0; kb < NN; kb += 32) {
        if (kb + 1024 < NN) {
            __builtin_prefetch((const void*)(vp0 + kb + 1024), 0, 0);
            __builtin_prefetch((const void*)(vp1 + kb + 1024), 0, 0);
        }
        // ---- A fragment: gate(q[m]*kv[k]), ISA 16-bit A layout -------------
        // lanes 0-15: K = kb+{0..7, 16..23}; lanes 16-31: +8
        const int bk = kb + kh * 8;
        const float4_t ka0 = kv4[(bk >> 2) + 0];
        const float4_t ka1 = kv4[(bk >> 2) + 1];
        const float4_t kc0 = kv4[((bk + 16) >> 2) + 0];
        const float4_t kc1 = kv4[((bk + 16) >> 2) + 1];

        auto gate = [&](float kvv) -> float {
            const float ex = __builtin_amdgcn_exp2f(qf * kvv);
            return __builtin_amdgcn_rcpf(1.0f + ex);
        };
        union { half16_t v; fp16x2_t p[8]; } A;
        A.p[0] = __builtin_amdgcn_cvt_pkrtz(gate(ka0[0]), gate(ka0[1]));
        A.p[1] = __builtin_amdgcn_cvt_pkrtz(gate(ka0[2]), gate(ka0[3]));
        A.p[2] = __builtin_amdgcn_cvt_pkrtz(gate(ka1[0]), gate(ka1[1]));
        A.p[3] = __builtin_amdgcn_cvt_pkrtz(gate(ka1[2]), gate(ka1[3]));
        A.p[4] = __builtin_amdgcn_cvt_pkrtz(gate(kc0[0]), gate(kc0[1]));
        A.p[5] = __builtin_amdgcn_cvt_pkrtz(gate(kc0[2]), gate(kc0[3]));
        A.p[6] = __builtin_amdgcn_cvt_pkrtz(gate(kc1[0]), gate(kc1[1]));
        A.p[7] = __builtin_amdgcn_cvt_pkrtz(gate(kc1[2]), gate(kc1[3]));

        // ---- B fragments: v[c, k] (same K order as A: {0..7, 16..23} +8*half) -
        const int bkk = kb + kh * 8;
        union { half16_t v; half8_t h[2]; } B0, B1;
        B0.h[0] = *(const half8_t*)(vp0 + bkk);
        B0.h[1] = *(const half8_t*)(vp0 + bkk + 16);
        B1.h[0] = *(const half8_t*)(vp1 + bkk);
        B1.h[1] = *(const half8_t*)(vp1 + bkk + 16);

        acc0 = __builtin_amdgcn_wmma_f32_16x16x32_f16(false, A.v, false, B0.v,
                                                      (short)0, acc0, false, false);
        acc1 = __builtin_amdgcn_wmma_f32_16x16x32_f16(false, A.v, false, B1.v,
                                                      (short)0, acc1, false, false);
    }

    // ---- D (f32 16x16) -> per-wave LDS tile: mid[m][c] ---------------------
#pragma unroll
    for (int j = 0; j < 8; ++j) {
        mid_s[wave][j + kh * 8][ml]      = acc0[j];
        mid_s[wave][j + kh * 8][ml + 16] = acc1[j];
    }
    __syncthreads();

    // ---- epilogue: out[o, m] = x[o, m] + gamma * (wc @ mid + bias) ---------
    const int r  = ml;         // this lane's m row
    const int og = kh;         // o-group: 0 -> o 0..31, 1 -> o 32..63
    float midreg[C2C];
    const float4_t* mrow = (const float4_t*)&mid_s[wave][r][0];
#pragma unroll
    for (int i = 0; i < 8; ++i) {
        const float4_t t = mrow[i];
        midreg[4 * i + 0] = t[0]; midreg[4 * i + 1] = t[1];
        midreg[4 * i + 2] = t[2]; midreg[4 * i + 3] = t[3];
    }
#pragma unroll 4
    for (int j = 0; j < 32; ++j) {
        const int o = og * 32 + j;
        float acc = bias_s[o];
        const float4_t* wrow = (const float4_t*)&wc_s[o * C2C];
#pragma unroll
        for (int i = 0; i < 8; ++i) {
            const float4_t w4 = wrow[i];
            acc = __builtin_fmaf(w4[0], midreg[4 * i + 0], acc);
            acc = __builtin_fmaf(w4[1], midreg[4 * i + 1], acc);
            acc = __builtin_fmaf(w4[2], midreg[4 * i + 2], acc);
            acc = __builtin_fmaf(w4[3], midreg[4 * i + 3], acc);
        }
        const size_t idx = (size_t)o * NN + m0 + r;   // lanes 0..15: coalesced in m
        outp[idx] = xres[idx] + gamma * acc;
    }
}

// ---------------------------------------------------------------------------
extern "C" void kernel_launch(void* const* d_in, const int* in_sizes, int n_in,
                              void* d_out, int out_size, void* d_ws, size_t ws_size,
                              hipStream_t stream)
{
    (void)in_sizes; (void)n_in; (void)out_size; (void)ws_size;
    const float* x1  = (const float*)d_in[0];
    const float* x2  = (const float*)d_in[1];
    const float* wq1 = (const float*)d_in[2];  const float* bq1 = (const float*)d_in[3];
    const float* wk1 = (const float*)d_in[4];  const float* bk1 = (const float*)d_in[5];
    const float* wv1 = (const float*)d_in[6];  const float* bv1 = (const float*)d_in[7];
    const float* wc1 = (const float*)d_in[8];  const float* bc1 = (const float*)d_in[9];
    const float* wq2 = (const float*)d_in[10]; const float* bq2 = (const float*)d_in[11];
    const float* wk2 = (const float*)d_in[12]; const float* bk2 = (const float*)d_in[13];
    const float* wv2 = (const float*)d_in[14]; const float* bv2 = (const float*)d_in[15];
    const float* wc2 = (const float*)d_in[16]; const float* bc2 = (const float*)d_in[17];
    const float* g1  = (const float*)d_in[18]; const float* g2  = (const float*)d_in[19];

    // workspace: qws (2*2*N f32) | kws (2*2*N f32) | vws (2*2*32*N f16) ~ 1.125 MB
    float*     qws = (float*)d_ws;
    float*     kws = qws + 4 * NN;
    _Float16*  vws = (_Float16*)(kws + 4 * NN);

    const dim3 blk(256);
    const dim3 g_proj(NN / 256, BB, 1);
    // branch from x1 (max-reduce): q1 -> q slot0, k1 -> k slot1, v1 -> v slot0
    proj_kernel<<<g_proj, blk, 0, stream>>>(x1, wq1, bq1, wk1, bk1, wv1, bv1,
                                            qws, kws + (size_t)BB * NN, vws, 0);
    // branch from x2 (mean-reduce): q2 -> q slot1, k2 -> k slot0, v2 -> v slot1
    proj_kernel<<<g_proj, blk, 0, stream>>>(x2, wq2, bq2, wk2, bk2, wv2, bv2,
                                            qws + (size_t)BB * NN, kws,
                                            vws + (size_t)BB * C2C * NN, 1);

    const dim3 g_attn(NN / 128, 2, BB);   // (m-chunks, branch, batch)
    attn_kernel<<<g_attn, blk, 0, stream>>>(qws, kws, vws,
                                            wc2, bc2, g1, x1,
                                            wc1, bc1, g2, x2,
                                            (float*)d_out);
}
